// MultiHeadAttention_9509057593852
// MI455X (gfx1250) — matrix operand order repaired and checked by reading, not compiled, not run on hardware
//
#include <hip/hip_runtime.h>
#include <hip/hip_bf16.h>

// ---------------------------------------------------------------------------
// MI455X (gfx1250) fused multi-head attention, bf16 WMMA + f32 accumulate.
// B=2, S=2048, D=1024, H=16, dh=64.  Flash-style fusion keeps the 512MB
// logits matrix out of HBM; GEMMs run on v_wmma_f32_16x16x32_bf16; K/V tile
// staging uses the Tensor Data Mover (tensor_load_to_lds + s_wait_tensorcnt).
// ---------------------------------------------------------------------------

typedef __bf16 bf16;
typedef __attribute__((ext_vector_type(16))) __bf16 v16bf;
typedef __attribute__((ext_vector_type(8)))  float  v8f;
typedef __attribute__((ext_vector_type(4)))  unsigned int v4u;
typedef __attribute__((ext_vector_type(8)))  int v8i;
typedef __attribute__((ext_vector_type(4)))  int v4i;

#define CB 2
#define CS 2048
#define CD 1024
#define CH 16
#define CDH 64

#if __has_builtin(__builtin_amdgcn_tensor_load_to_lds)
#define USE_TDM 1
#endif

// ---------------------------------------------------------------------------
// Tiled GEMM: C[M,N] = A[M,K] * W[K,N].  Block tile 256x64, 8 waves; each
// wave computes 32x64 (2 M-subtiles x 4 N-tiles = 8 accumulators), K stepped
// by 32 through LDS.  B-fragments are reused across both M-subtiles.
// OUT_MODE 0: fp32 row-major out.  OUT_MODE 1: bf16 scattered to [B,H,S,dh].
// ---------------------------------------------------------------------------
template <bool A_BF16, int OUT_MODE>
__global__ __launch_bounds__(256) void gemm_kernel(const void* __restrict__ Aptr,
                                                   const float* __restrict__ W,
                                                   void* __restrict__ Out,
                                                   int M, int N, int K) {
  __shared__ bf16 lds_a[256][36];  // +4 pad vs bank conflicts (18KB)
  __shared__ bf16 lds_b[32][68];

  const int t = threadIdx.x;
  const int wave = t >> 5;
  const int lane = t & 31;
  const int n0 = blockIdx.x * 64;
  const int m0 = blockIdx.y * 256;

  v8f acc[2][4] = {};

  for (int k0 = 0; k0 < K; k0 += 32) {
    __syncthreads();
    // ---- stage A tile (256x32), fp32 -> bf16 on the fly ----
    for (int i = t; i < 256 * 32; i += 256) {
      const int r = i >> 5, c = i & 31;
      float av;
      if constexpr (A_BF16)
        av = (float)((const bf16*)Aptr)[(size_t)(m0 + r) * K + k0 + c];
      else
        av = ((const float*)Aptr)[(size_t)(m0 + r) * K + k0 + c];
      lds_a[r][c] = (bf16)av;
    }
    // ---- stage B tile (32x64) ----
    for (int i = t; i < 32 * 64; i += 256) {
      const int r = i >> 6, c = i & 63;
      lds_b[r][c] = (bf16)W[(size_t)(k0 + r) * N + n0 + c];
    }
    // gfx1250 prefetch of next tiles (global_prefetch_b8)
    if (k0 + 32 < K && lane == 0) {
      __builtin_prefetch(&W[(size_t)(k0 + 32 + wave * 4) * N + n0], 0, 1);
      if constexpr (!A_BF16)
        __builtin_prefetch(&((const float*)Aptr)[(size_t)(m0 + wave * 32) * K + k0 + 32], 0, 1);
    }
    __syncthreads();

    // ---- A fragments (2 M-subtiles): contiguous k = 16*(lane/16) + e per lane ----
    const int agrp = (lane >> 4) << 4;  // +16 K for lanes 16..31
    v16bf af[2];
#pragma unroll
    for (int mt = 0; mt < 2; ++mt) {
      const int arow = wave * 32 + mt * 16 + (lane & 15);
#pragma unroll
      for (int e = 0; e < 16; ++e) {
        const int v = e >> 1, p = e & 1;
        const int kk = agrp + e;
        af[mt][e] = lds_a[arow][kk];
      }
    }
    // ---- 4 B-frags x 2 M-subtiles = 8 WMMAs per K step ----
    const int bcolb = lane & 15;
    const int bkb = (lane >> 4) << 4;  // K base 0 / 16
#pragma unroll
    for (int n = 0; n < 4; ++n) {
      v16bf bfrag;
      const int bcol = n * 16 + bcolb;
#pragma unroll
      for (int e = 0; e < 16; ++e) bfrag[e] = lds_b[bkb + e][bcol];
#pragma unroll
      for (int mt = 0; mt < 2; ++mt)
        acc[mt][n] = __builtin_amdgcn_wmma_f32_16x16x32_bf16(
            false, af[mt], false, bfrag, (short)0, acc[mt][n], false, false);
    }
  }

  // ---- writeback: C layout VGPR r -> M = r + 8*(lane>=16), N = lane%16 ----
  const int colb = lane & 15;
  const int rowadd = (lane >> 4) << 3;
#pragma unroll
  for (int mt = 0; mt < 2; ++mt) {
#pragma unroll
    for (int n = 0; n < 4; ++n) {
#pragma unroll
      for (int r = 0; r < 8; ++r) {
        const int row = m0 + wave * 32 + mt * 16 + r + rowadd;
        const int col = n0 + n * 16 + colb;
        const float val = acc[mt][n][r];
        if constexpr (OUT_MODE == 0) {
          ((float*)Out)[(size_t)row * N + col] = val;
        } else {
          const int b = row >> 11;          // / S
          const int s = row & (CS - 1);
          const int h = col >> 6;           // / dh
          const int dd = col & (CDH - 1);
          ((bf16*)Out)[(((size_t)(b * CH + h) * CS + s) << 6) + dd] = (bf16)val;
        }
      }
    }
  }
}

// ---------------------------------------------------------------------------
// K = Q / ||Q|| over dh (Reformer shared-QK).  One thread per (b,h,s) row.
// ---------------------------------------------------------------------------
__global__ __launch_bounds__(256) void norm_kernel(const bf16* __restrict__ Q,
                                                   bf16* __restrict__ Kout,
                                                   int total) {
  const int i = blockIdx.x * blockDim.x + threadIdx.x;
  if (i >= total) return;
  const bf16* q = Q + (size_t)i * CDH;
  float ss = 0.f;
#pragma unroll 8
  for (int d = 0; d < CDH; ++d) {
    const float x = (float)q[d];
    ss += x * x;
  }
  const float inv = rsqrtf(ss + 1e-12f);
  bf16* o = Kout + (size_t)i * CDH;
#pragma unroll 8
  for (int d = 0; d < CDH; ++d) o[d] = (bf16)((float)q[d] * inv);
}

// Deterministic dropout hash (compile-only stand-in for jax PRNG).
__device__ __forceinline__ float drop_scale(unsigned bh, unsigned qr, unsigned kv) {
  unsigned s = (bh << 22) ^ (qr << 11) ^ kv ^ 0x9E3779B9u;
  s = (s ^ 61u) ^ (s >> 16);
  s *= 9u;
  s ^= s >> 4;
  s *= 0x27d4eb2du;
  s ^= s >> 15;
  return (s & 0xFFFFFFu) < 15099494u ? (1.0f / 0.9f) : 0.0f;  // keep-prob 0.9
}

#ifdef USE_TDM
// Issue a TDM 2-D tile load: 32 rows x 64 bf16, row stride 64 elements in
// memory, +8 elements LDS row padding via the descriptor's pad fields
// (pad_interval: 32 DWORDs -> enum 4, pad_amount: 4 DWORDs -> enum 3).
// 6-arg builtin form: (g0, g1, g2, g3, g4, cpol); groups 2..4 zero (2-D).
__device__ __forceinline__ void tdm_load_tile(unsigned lds_addr, const bf16* gsrc) {
  const unsigned long long ga = (unsigned long long)(size_t)gsrc;
  v4u g0;
  g0[0] = 1u;                                   // count=1, user mode
  g0[1] = lds_addr;                             // lds_addr (bytes)
  g0[2] = (unsigned)ga;                         // global_addr[31:0]
  g0[3] = (unsigned)((ga >> 32) & 0x01FFFFFFull) | 0x80000000u;  // addr[56:32], type=2
  v8i g1;
  g1[0] = 0x07110000;      // data_size=2B, pad_enable, pad_interval=32dw, pad_amount=4dw
  g1[1] = (CDH << 16);     // tensor_dim0[15:0]=64 at bits 63:48
  g1[2] = (CS << 16);      // tensor_dim0 hi=0 | tensor_dim1[15:0]=2048
  g1[3] = (CDH << 16);     // tensor_dim1 hi=0 | tile_dim0=64 at bits 127:112
  g1[4] = 32;              // tile_dim1=32, tile_dim2=0
  g1[5] = CDH;             // tensor_dim0_stride=64 (data_size units)
  g1[6] = 0;
  g1[7] = 0;
  v4i z4 = {};
  v8i z8 = {};
  __builtin_amdgcn_tensor_load_to_lds(g0, g1, z4, z4, z8, 0);
}
#endif

// ---------------------------------------------------------------------------
// Flash attention: grid (S/128, B*H).  Block = 8 waves; each wave owns 16 q
// rows x full dh=64.  kv streamed in 32-row tiles via TDM (or manual LDS
// staging); online softmax with 16-lane butterfly reductions matching the
// WMMA C-row layout.
// ---------------------------------------------------------------------------
__global__ __launch_bounds__(256) void flash_kernel(const bf16* __restrict__ Q,
                                                    const bf16* __restrict__ Kn,
                                                    const bf16* __restrict__ V,
                                                    bf16* __restrict__ Merged) {
  __shared__ bf16 lds_k[32][72];
  __shared__ bf16 lds_v[32][72];
  __shared__ bf16 lds_p[8][16][36];  // per-wave P staging (C-layout -> A-layout)

  const int t = threadIdx.x;
  const int wave = t >> 5;
  const int lane = t & 31;
  const int qblk = blockIdx.x;
  const int bh = blockIdx.y;  // = b*H + h
  const int b = bh >> 4;
  const int h = bh & 15;
  const int q0 = qblk * 128;
  const size_t head_off = (size_t)bh * CS * CDH;

  // ---- preload Q fragments (16 rows x 64 dh) -> two 16x32 A-frags ----
  const int qrow = q0 + wave * 16 + (lane & 15);
  const int agrp = (lane >> 4) << 4;
  v16bf qf[2];
#pragma unroll
  for (int f = 0; f < 2; ++f) {
#pragma unroll
    for (int e = 0; e < 16; ++e) {
      const int v = e >> 1, p = e & 1;
      const int kk = 32 * f + agrp + e;
      qf[f][e] = Q[head_off + (size_t)qrow * CDH + kk];
    }
  }

  float mrow[8], lrow[8];
#pragma unroll
  for (int r = 0; r < 8; ++r) {
    mrow[r] = -1e30f;
    lrow[r] = 0.f;
  }
  v8f o[4] = {};

  const float scale = 0.125f;  // 1/sqrt(dh)
  const int rowadd = (lane >> 4) << 3;
  const int colb = lane & 15;
  const int kmax = q0 + 128;  // causal bound for this q block

  for (int kb = 0; kb < kmax; kb += 32) {
    __syncthreads();
#ifdef USE_TDM
    if (wave == 0) {
      tdm_load_tile((unsigned)(size_t)&lds_k[0][0], Kn + head_off + (size_t)kb * CDH);
      tdm_load_tile((unsigned)(size_t)&lds_v[0][0], V + head_off + (size_t)kb * CDH);
      __builtin_amdgcn_s_wait_tensorcnt(0);
    }
#else
    for (int i = t; i < 32 * 64; i += 256) {
      const int r = i >> 6, c = i & 63;
      lds_k[r][c] = Kn[head_off + (size_t)(kb + r) * CDH + c];
      lds_v[r][c] = V[head_off + (size_t)(kb + r) * CDH + c];
    }
#endif
    __syncthreads();

    // ---- scores: two 16x16 tiles (kv 0..15 / 16..31), K-dim = dh = 2 steps ----
    v8f sc[2];
#pragma unroll
    for (int c = 0; c < 2; ++c) {
      v8f s_ = {};
      const int j = c * 16 + colb;  // kv row within tile (B-matrix N)
#pragma unroll
      for (int f = 0; f < 2; ++f) {
        v16bf bfrag;
        const int db = 32 * f + ((lane >> 4) << 4);
#pragma unroll
        for (int e = 0; e < 16; ++e) bfrag[e] = lds_k[j][db + e];
        s_ = __builtin_amdgcn_wmma_f32_16x16x32_bf16(
            false, qf[f], false, bfrag, (short)0, s_, false, false);
      }
      sc[c] = s_;
    }

    // ---- online softmax per row + P staging with dropout ----
    const int kv0 = kb + colb;
    const int kv1 = kb + 16 + colb;
#pragma unroll
    for (int r = 0; r < 8; ++r) {
      const int qr = q0 + wave * 16 + r + rowadd;
      float s0 = sc[0][r] * scale;
      float s1 = sc[1][r] * scale;
      if (kv0 > qr) s0 = -1e9f;  // causal look-ahead mask
      if (kv1 > qr) s1 = -1e9f;
      float x = fmaxf(s0, s1);
#pragma unroll
      for (int off = 1; off < 16; off <<= 1) x = fmaxf(x, __shfl_xor(x, off, 32));
      const float m_new = fmaxf(mrow[r], x);
      const float ef = __expf(mrow[r] - m_new);
      const float p0 = __expf(s0 - m_new);
      const float p1 = __expf(s1 - m_new);
      float rs = p0 + p1;
#pragma unroll
      for (int off = 1; off < 16; off <<= 1) rs += __shfl_xor(rs, off, 32);
      lrow[r] = lrow[r] * ef + rs;
      mrow[r] = m_new;
#pragma unroll
      for (int n = 0; n < 4; ++n) o[n][r] *= ef;
      // dropout applied to the numerator only (denominator uses undropped p)
      lds_p[wave][r + rowadd][colb] = (bf16)(p0 * drop_scale(bh, qr, kv0));
      lds_p[wave][r + rowadd][16 + colb] = (bf16)(p1 * drop_scale(bh, qr, kv1));
    }

    // intra-wave cross-lane RAW through LDS: drain DS ops (gfx1250 split counter)
    asm volatile("s_wait_dscnt 0" ::: "memory");

    // ---- P back as 16x32 A-fragment ----
    v16bf pf;
    const int prow = lane & 15;
#pragma unroll
    for (int e = 0; e < 16; ++e) {
      const int v = e >> 1, p = e & 1;
      const int kk = agrp + e;
      pf[e] = lds_p[wave][prow][kk];
    }

    // ---- O += P @ V  (4 dh tiles) ----
    const int kvb = (lane >> 4) << 4;
#pragma unroll
    for (int n = 0; n < 4; ++n) {
      v16bf vfrag;
      const int dcol = n * 16 + colb;
#pragma unroll
      for (int e = 0; e < 16; ++e) vfrag[e] = lds_v[kvb + e][dcol];
      o[n] = __builtin_amdgcn_wmma_f32_16x16x32_bf16(
          false, pf, false, vfrag, (short)0, o[n], false, false);
    }
  }

  // ---- normalize by l and write merged [B,S,D] ----
#pragma unroll
  for (int r = 0; r < 8; ++r) {
    const int qr = q0 + wave * 16 + r + rowadd;
    const float invl = 1.0f / lrow[r];
#pragma unroll
    for (int n = 0; n < 4; ++n) {
      const int dcol = n * 16 + colb;
      Merged[(((size_t)(b * CS + qr)) << 10) + h * CDH + dcol] = (bf16)(o[n][r] * invl);
    }
  }
}

// ---------------------------------------------------------------------------
extern "C" void kernel_launch(void* const* d_in, const int* in_sizes, int n_in,
                              void* d_out, int out_size, void* d_ws, size_t ws_size,
                              hipStream_t stream) {
  const float* qk  = (const float*)d_in[0];
  const float* v   = (const float*)d_in[1];
  const float* Wqk = (const float*)d_in[2];
  const float* Wv  = (const float*)d_in[3];
  const float* Wo  = (const float*)d_in[4];
  float* out = (float*)d_out;

  const size_t elems = (size_t)CB * CS * CD;  // 4,194,304
  bf16* Qbf = (bf16*)d_ws;
  bf16* Kbf = Qbf + elems;
  bf16* Vbf = Kbf + elems;
  bf16* Mbf = Vbf + elems;

  const int M = CB * CS;  // 4096
  dim3 blk(256);
  dim3 gGemm(CD / 64, M / 256);  // (16, 16)

  // Projections into per-head bf16 layout
  gemm_kernel<false, 1><<<gGemm, blk, 0, stream>>>(qk, Wqk, Qbf, M, CD, CD);
  gemm_kernel<false, 1><<<gGemm, blk, 0, stream>>>(v, Wv, Vbf, M, CD, CD);
  // Shared-QK key normalization
  const int rows = CB * CH * CS;  // 65536
  norm_kernel<<<(rows + 255) / 256, 256, 0, stream>>>(Qbf, Kbf, rows);
  // Fused flash attention -> merged [B,S,D] bf16
  flash_kernel<<<dim3(CS / 128, CB * CH), blk, 0, stream>>>(Qbf, Kbf, Vbf, Mbf);
  // Output projection -> fp32
  gemm_kernel<true, 0><<<gGemm, blk, 0, stream>>>(Mbf, Wo, out, M, CD, CD);
}
